// MoE_70798240907976
// MI455X (gfx1250) — matrix operand order repaired and checked by reading, not compiled, not run on hardware
//
#include <hip/hip_runtime.h>
#include <hip/hip_bf16.h>

// ---------------- problem constants ----------------
#define T_TOK 4096   // B*S = 2*2048
#define EMB   1024
#define FF    2048
#define NE    8
// top-k = 2

// ---------------- tile constants -------------------
#define BM 128       // block rows (token-expert rows)
#define BN 64        // block cols
#define BK 32        // k per WMMA stage
#define LDP 40       // padded LDS row (bf16 elems): 80B rows -> conflict-free b128 reads

typedef __attribute__((ext_vector_type(16))) __bf16 v16bf;
typedef __attribute__((ext_vector_type(8)))  __bf16 bf16x8;
typedef __attribute__((ext_vector_type(4)))  __bf16 bf16x4;
typedef __attribute__((ext_vector_type(2)))  __bf16 bf16x2;
typedef __attribute__((ext_vector_type(8)))  float  v8f;

union FragBF { v16bf v; bf16x8 h[2]; };

__device__ __forceinline__ float silu_f(float v) { return v / (1.0f + __expf(-v)); }

// ---------------------------------------------------------------------------
// Kernel 0: convert x (fp32) -> bf16 once
// ---------------------------------------------------------------------------
__global__ __launch_bounds__(256) void moe_cvt_x(const float* __restrict__ x,
                                                 __bf16* __restrict__ xb) {
  int i = (blockIdx.x * 256 + threadIdx.x) * 4;
  float4 v = *(const float4*)(x + i);
  bf16x4 o;
  o[0] = (__bf16)v.x; o[1] = (__bf16)v.y; o[2] = (__bf16)v.z; o[3] = (__bf16)v.w;
  *(bf16x4*)(xb + i) = o;
}

// ---------------------------------------------------------------------------
// Kernel 1: router — logits, softmax, top-2 (raw softmax probs, as reference)
// ---------------------------------------------------------------------------
__global__ __launch_bounds__(128) void moe_router(const float* __restrict__ x,
                                                  const float* __restrict__ rw,
                                                  float* __restrict__ top2w,
                                                  int* __restrict__ top2i) {
  __shared__ float rws[NE * EMB];                 // 32 KB
  for (int i = threadIdx.x; i < NE * EMB; i += 128) rws[i] = rw[i];
  __syncthreads();

  int t = blockIdx.x * 128 + threadIdx.x;
  const float* xr = x + (size_t)t * EMB;
  float acc[NE];
#pragma unroll
  for (int e = 0; e < NE; ++e) acc[e] = 0.f;
  for (int d = 0; d < EMB; d += 4) {
    float4 xv = *(const float4*)(xr + d);
#pragma unroll
    for (int e = 0; e < NE; ++e) {
      float4 wv = *(const float4*)(&rws[e * EMB + d]);   // LDS broadcast
      acc[e] += xv.x * wv.x + xv.y * wv.y + xv.z * wv.z + xv.w * wv.w;
    }
  }
  float mx = acc[0];
#pragma unroll
  for (int e = 1; e < NE; ++e) mx = fmaxf(mx, acc[e]);
  float p[NE]; float s = 0.f;
#pragma unroll
  for (int e = 0; e < NE; ++e) { p[e] = __expf(acc[e] - mx); s += p[e]; }
  float inv = 1.f / s;
#pragma unroll
  for (int e = 0; e < NE; ++e) p[e] *= inv;

  int i1 = 0; float p1 = p[0];
#pragma unroll
  for (int e = 1; e < NE; ++e) if (p[e] > p1) { p1 = p[e]; i1 = e; }
  int i2 = -1; float p2 = -1.f;
#pragma unroll
  for (int e = 0; e < NE; ++e) if (e != i1 && p[e] > p2) { p2 = p[e]; i2 = e; }

  top2i[2 * t] = i1; top2i[2 * t + 1] = i2;
  top2w[2 * t] = p1; top2w[2 * t + 1] = p2;
}

// ---------------------------------------------------------------------------
// Kernel 2: deterministic placement (8 waves, one expert each; ballot scans)
// ---------------------------------------------------------------------------
__global__ __launch_bounds__(256) void moe_place(const int* __restrict__ top2i,
                                                 int* __restrict__ row_token,
                                                 int* __restrict__ token_slot,
                                                 int* __restrict__ counts,
                                                 int* __restrict__ offs) {
  __shared__ int cnt_s[NE];
  __shared__ int off_s[NE + 1];
  int lane = threadIdx.x & 31;
  int e = threadIdx.x >> 5;

  int cnt = 0;
  for (int base = 0; base < T_TOK; base += 32) {
    int t = base + lane;
    int a = top2i[2 * t], b = top2i[2 * t + 1];
    bool ch = (a == e) || (b == e);
    unsigned mask = __builtin_amdgcn_ballot_w32(ch);
    cnt += __builtin_popcount(mask);
  }
  if (lane == 0) cnt_s[e] = cnt;
  __syncthreads();
  if (threadIdx.x == 0) {
    int o = 0;
    for (int i = 0; i < NE; ++i) { off_s[i] = o; o += cnt_s[i]; counts[i] = cnt_s[i]; }
    off_s[NE] = o;
    for (int i = 0; i <= NE; ++i) offs[i] = off_s[i];
  }
  __syncthreads();

  int pos = off_s[e];
  unsigned below = (1u << lane) - 1u;
  for (int base = 0; base < T_TOK; base += 32) {
    int t = base + lane;
    int a = top2i[2 * t], b = top2i[2 * t + 1];
    bool ch = (a == e) || (b == e);
    int k = (a == e) ? 0 : 1;
    unsigned mask = __builtin_amdgcn_ballot_w32(ch);
    int my = __builtin_popcount(mask & below);
    if (ch) {
      int p2 = pos + my;
      row_token[p2] = t;
      token_slot[2 * t + k] = p2;
    }
    pos += __builtin_popcount(mask);
  }
}

// ----- WMMA fragment loaders (CDNA5 16x16x32 bf16 VGPR layouts) -----
__device__ __forceinline__ v16bf load_a_frag(const __bf16* row_base, int lane) {
  int kh = lane >> 4;
  FragBF f;
  f.h[0] = *(const bf16x8*)(row_base + 8 * kh);
  f.h[1] = *(const bf16x8*)(row_base + 16 + 8 * kh);
  return f.v;
}
__device__ __forceinline__ v16bf load_b_frag(const __bf16* row_base, int lane) {
  int kh = lane >> 4;
  FragBF f;
  f.h[0] = *(const bf16x8*)(row_base + 8 * kh);
  f.h[1] = *(const bf16x8*)(row_base + 16 + 8 * kh);
  return f.v;
}

// Vectorized weight staging: thread loads float4 from two adjacent k-rows,
// packs vertical bf16x2 pairs -> b32 LDS stores into transposed [n][k] tile.
__device__ __forceinline__ void stage_w_tile(const float* __restrict__ wrow0,
                                             int row_stride, __bf16* __restrict__ Ws,
                                             int dp, int f0) {
  float4 r0 = *(const float4*)(wrow0 + (size_t)(2 * dp) * row_stride + f0);
  float4 r1 = *(const float4*)(wrow0 + (size_t)(2 * dp + 1) * row_stride + f0);
  bf16x2 p;
  p[0] = (__bf16)r0.x; p[1] = (__bf16)r1.x; *(bf16x2*)(&Ws[(f0 + 0) * LDP + 2 * dp]) = p;
  p[0] = (__bf16)r0.y; p[1] = (__bf16)r1.y; *(bf16x2*)(&Ws[(f0 + 1) * LDP + 2 * dp]) = p;
  p[0] = (__bf16)r0.z; p[1] = (__bf16)r1.z; *(bf16x2*)(&Ws[(f0 + 2) * LDP + 2 * dp]) = p;
  p[0] = (__bf16)r0.w; p[1] = (__bf16)r1.w; *(bf16x2*)(&Ws[(f0 + 3) * LDP + 2 * dp]) = p;
}

// ---------------------------------------------------------------------------
// Kernel 3: grouped gate/up GEMM + fused SiLU-gate -> H (bf16).
// grid = (T/BM, FF/BN, E). Block tile 128x64; 8 waves as 4x2; each wave does
// 32x32 per GEMM: 2 A-frags shared by 4 B-frags -> 8 WMMAs per k-step.
// ---------------------------------------------------------------------------
__global__ __launch_bounds__(256) void moe_gate_up(const __bf16* __restrict__ xb,
                                                   const float* __restrict__ gw,
                                                   const float* __restrict__ uw,
                                                   const int* __restrict__ counts,
                                                   const int* __restrict__ offs,
                                                   const int* __restrict__ row_token,
                                                   __bf16* __restrict__ H) {
  int e = blockIdx.z;
  int cnt = counts[e];
  int m0 = blockIdx.x * BM;
  if (m0 >= cnt) return;
  int base = offs[e];
  int n0 = blockIdx.y * BN;
  int tid = threadIdx.x;
  int lane = tid & 31, wave = tid >> 5;
  int wm = wave >> 1, wn = wave & 1;    // wm 0..3 (32 rows), wn 0..1 (32 cols)

  __shared__ __bf16 Xs[BM * LDP];
  __shared__ __bf16 Gs[BN * LDP];
  __shared__ __bf16 Us[BN * LDP];
  __shared__ int rtok[BM];

  for (int i = tid; i < BM; i += 256)
    rtok[i] = (m0 + i < cnt) ? row_token[base + m0 + i] : 0;
  __syncthreads();

  int mrows = cnt - m0; if (mrows > BM) mrows = BM;

  // hoisted gather state: loop-invariant source pointers for the X staging
  int m_a = tid >> 2, c_a = (tid & 3) * 8;
  bool va0 = m_a < mrows, va1 = (m_a + 64) < mrows;
  const __bf16* xsrc0 = xb + (size_t)rtok[m_a] * EMB + c_a;
  const __bf16* xsrc1 = xb + (size_t)rtok[m_a + 64] * EMB + c_a;
  {  // zero-fill padding rows ONCE (they are never overwritten)
    bf16x8 z;
#pragma unroll
    for (int j = 0; j < 8; ++j) z[j] = (__bf16)0.f;
    if (!va0) *(bf16x8*)(&Xs[m_a * LDP + c_a]) = z;
    if (!va1) *(bf16x8*)(&Xs[(m_a + 64) * LDP + c_a]) = z;
  }

  v8f cg[2][2] = {{{}, {}}, {{}, {}}};
  v8f cu[2][2] = {{{}, {}}, {{}, {}}};

  int dp = tid >> 4;            // 0..15 -> k-rows 2dp, 2dp+1
  int f0 = (tid & 15) * 4;      // 0..60 -> 4 n-cols
  const float* gbase0 = gw + ((size_t)e * EMB) * FF + n0;
  const float* ubase0 = uw + ((size_t)e * EMB) * FF + n0;

  for (int dk = 0; dk < EMB; dk += BK) {
    __syncthreads();
    // stage gathered X rows (pointers hoisted; only +dk varies)
    if (va0) *(bf16x8*)(&Xs[m_a * LDP + c_a]) = *(const bf16x8*)(xsrc0 + dk);
    if (va1) *(bf16x8*)(&Xs[(m_a + 64) * LDP + c_a]) = *(const bf16x8*)(xsrc1 + dk);
    // stage gate/up tiles transposed (vectorized fp32->bf16)
    const float* gbase = gbase0 + (size_t)dk * FF;
    const float* ubase = ubase0 + (size_t)dk * FF;
    stage_w_tile(gbase, FF, Gs, dp, f0);
    stage_w_tile(ubase, FF, Us, dp, f0);
    if (dk + BK < EMB) {   // prefetch next k-tile (global_prefetch_b8)
      __builtin_prefetch(gbase + (size_t)(BK + 2 * dp) * FF + f0, 0, 1);
      __builtin_prefetch(ubase + (size_t)(BK + 2 * dp) * FF + f0, 0, 1);
    }
    __syncthreads();

    v16bf a0 = load_a_frag(&Xs[(wm * 32 + (lane & 15)) * LDP], lane);
    v16bf a1 = load_a_frag(&Xs[(wm * 32 + 16 + (lane & 15)) * LDP], lane);
    v16bf bg0 = load_b_frag(&Gs[(wn * 32 + (lane & 15)) * LDP], lane);
    v16bf bg1 = load_b_frag(&Gs[(wn * 32 + 16 + (lane & 15)) * LDP], lane);
    v16bf bu0 = load_b_frag(&Us[(wn * 32 + (lane & 15)) * LDP], lane);
    v16bf bu1 = load_b_frag(&Us[(wn * 32 + 16 + (lane & 15)) * LDP], lane);

    cg[0][0] = __builtin_amdgcn_wmma_f32_16x16x32_bf16(false, a0, false, bg0, (short)0, cg[0][0], false, false);
    cg[0][1] = __builtin_amdgcn_wmma_f32_16x16x32_bf16(false, a0, false, bg1, (short)0, cg[0][1], false, false);
    cg[1][0] = __builtin_amdgcn_wmma_f32_16x16x32_bf16(false, a1, false, bg0, (short)0, cg[1][0], false, false);
    cg[1][1] = __builtin_amdgcn_wmma_f32_16x16x32_bf16(false, a1, false, bg1, (short)0, cg[1][1], false, false);
    cu[0][0] = __builtin_amdgcn_wmma_f32_16x16x32_bf16(false, a0, false, bu0, (short)0, cu[0][0], false, false);
    cu[0][1] = __builtin_amdgcn_wmma_f32_16x16x32_bf16(false, a0, false, bu1, (short)0, cu[0][1], false, false);
    cu[1][0] = __builtin_amdgcn_wmma_f32_16x16x32_bf16(false, a1, false, bu0, (short)0, cu[1][0], false, false);
    cu[1][1] = __builtin_amdgcn_wmma_f32_16x16x32_bf16(false, a1, false, bu1, (short)0, cu[1][1], false, false);
  }

  // epilogue: h = silu(g)*u -> bf16. C layout: m = vgpr + 8*(lane>=16), n = lane&15
  int kh = lane >> 4, n = lane & 15;
#pragma unroll
  for (int ai = 0; ai < 2; ++ai) {
#pragma unroll
    for (int v = 0; v < 8; ++v) {
      int gl = wm * 32 + ai * 16 + kh * 8 + v;
      if (m0 + gl < cnt) {
        size_t rowo = (size_t)(base + m0 + gl) * FF + n0 + wn * 32;
        H[rowo + n]      = (__bf16)(silu_f(cg[ai][0][v]) * cu[ai][0][v]);
        H[rowo + 16 + n] = (__bf16)(silu_f(cg[ai][1][v]) * cu[ai][1][v]);
      }
    }
  }
}

// ---------------------------------------------------------------------------
// Kernel 4: grouped down GEMM: O[row, :] = H[row, :] @ down_w[e]  (fp32 out)
// grid = (T/BM, EMB/BN, E). Wave tile 32x32 -> 4 WMMAs per k-step.
// ---------------------------------------------------------------------------
__global__ __launch_bounds__(256) void moe_down(const __bf16* __restrict__ H,
                                                const float* __restrict__ dw,
                                                const int* __restrict__ counts,
                                                const int* __restrict__ offs,
                                                float* __restrict__ O) {
  int e = blockIdx.z;
  int cnt = counts[e];
  int m0 = blockIdx.x * BM;
  if (m0 >= cnt) return;
  int base = offs[e];
  int n0 = blockIdx.y * BN;
  int tid = threadIdx.x;
  int lane = tid & 31, wave = tid >> 5;
  int wm = wave >> 1, wn = wave & 1;

  __shared__ __bf16 Hs[BM * LDP];
  __shared__ __bf16 Ds[BN * LDP];

  int mrows = cnt - m0; if (mrows > BM) mrows = BM;

  int m_a = tid >> 2, c_a = (tid & 3) * 8;
  bool va0 = m_a < mrows, va1 = (m_a + 64) < mrows;
  const __bf16* hsrc0 = H + (size_t)(base + m0 + m_a) * FF + c_a;
  const __bf16* hsrc1 = H + (size_t)(base + m0 + m_a + 64) * FF + c_a;
  {
    bf16x8 z;
#pragma unroll
    for (int j = 0; j < 8; ++j) z[j] = (__bf16)0.f;
    if (!va0) *(bf16x8*)(&Hs[m_a * LDP + c_a]) = z;
    if (!va1) *(bf16x8*)(&Hs[(m_a + 64) * LDP + c_a]) = z;
  }

  v8f c[2][2] = {{{}, {}}, {{}, {}}};

  int dp = tid >> 4;
  int f0 = (tid & 15) * 4;
  const float* dbase0 = dw + ((size_t)e * FF) * EMB + n0;

  for (int fk = 0; fk < FF; fk += BK) {
    __syncthreads();
    if (va0) *(bf16x8*)(&Hs[m_a * LDP + c_a]) = *(const bf16x8*)(hsrc0 + fk);
    if (va1) *(bf16x8*)(&Hs[(m_a + 64) * LDP + c_a]) = *(const bf16x8*)(hsrc1 + fk);
    const float* dbase = dbase0 + (size_t)fk * EMB;
    stage_w_tile(dbase, EMB, Ds, dp, f0);
    if (fk + BK < FF)
      __builtin_prefetch(dbase + (size_t)(BK + 2 * dp) * EMB + f0, 0, 1);
    __syncthreads();

    v16bf a0 = load_a_frag(&Hs[(wm * 32 + (lane & 15)) * LDP], lane);
    v16bf a1 = load_a_frag(&Hs[(wm * 32 + 16 + (lane & 15)) * LDP], lane);
    v16bf b0 = load_b_frag(&Ds[(wn * 32 + (lane & 15)) * LDP], lane);
    v16bf b1 = load_b_frag(&Ds[(wn * 32 + 16 + (lane & 15)) * LDP], lane);

    c[0][0] = __builtin_amdgcn_wmma_f32_16x16x32_bf16(false, a0, false, b0, (short)0, c[0][0], false, false);
    c[0][1] = __builtin_amdgcn_wmma_f32_16x16x32_bf16(false, a0, false, b1, (short)0, c[0][1], false, false);
    c[1][0] = __builtin_amdgcn_wmma_f32_16x16x32_bf16(false, a1, false, b0, (short)0, c[1][0], false, false);
    c[1][1] = __builtin_amdgcn_wmma_f32_16x16x32_bf16(false, a1, false, b1, (short)0, c[1][1], false, false);
  }

  int kh = lane >> 4, n = lane & 15;
#pragma unroll
  for (int ai = 0; ai < 2; ++ai) {
#pragma unroll
    for (int v = 0; v < 8; ++v) {
      int gl = wm * 32 + ai * 16 + kh * 8 + v;
      if (m0 + gl < cnt) {
        size_t rowo = (size_t)(base + m0 + gl) * EMB + n0 + wn * 32;
        O[rowo + n]      = c[ai][0][v];
        O[rowo + 16 + n] = c[ai][1][v];
      }
    }
  }
}

// ---------------------------------------------------------------------------
// Kernel 5: combine — out[t] = w0*O[slot0] + w1*O[slot1]   (float4 per thread)
// ---------------------------------------------------------------------------
__global__ __launch_bounds__(256) void moe_combine(const int* __restrict__ slot,
                                                   const float* __restrict__ w,
                                                   const float* __restrict__ O,
                                                   float* __restrict__ out) {
  int t = blockIdx.x;
  int s0 = slot[2 * t], s1 = slot[2 * t + 1];
  float w0 = w[2 * t], w1 = w[2 * t + 1];
  int d = threadIdx.x * 4;
  float4 a = *(const float4*)(O + (size_t)s0 * EMB + d);
  float4 b = *(const float4*)(O + (size_t)s1 * EMB + d);
  float4 r;
  r.x = w0 * a.x + w1 * b.x;
  r.y = w0 * a.y + w1 * b.y;
  r.z = w0 * a.z + w1 * b.z;
  r.w = w0 * a.w + w1 * b.w;
  *(float4*)(out + (size_t)t * EMB + d) = r;
}

// ---------------------------------------------------------------------------
extern "C" void kernel_launch(void* const* d_in, const int* in_sizes, int n_in,
                              void* d_out, int out_size, void* d_ws, size_t ws_size,
                              hipStream_t stream) {
  const float* x        = (const float*)d_in[0];
  const float* router_w = (const float*)d_in[1];
  const float* gate_w   = (const float*)d_in[2];
  const float* up_w     = (const float*)d_in[3];
  const float* down_w   = (const float*)d_in[4];
  float* out = (float*)d_out;

  char* ws = (char*)d_ws;
  size_t off = 0;
  auto carve = [&](size_t bytes) -> char* {
    char* p = ws + off;
    off += (bytes + 255) & ~(size_t)255;
    return p;
  };
  __bf16* xb       = (__bf16*)carve((size_t)T_TOK * EMB * sizeof(__bf16));          //  8 MB
  __bf16* Hbuf     = (__bf16*)carve((size_t)2 * T_TOK * FF * sizeof(__bf16));       // 32 MB
  float*  Obuf     = (float*) carve((size_t)2 * T_TOK * EMB * sizeof(float));       // 32 MB
  int*    top2i    = (int*)   carve((size_t)T_TOK * 2 * sizeof(int));
  float*  top2w    = (float*) carve((size_t)T_TOK * 2 * sizeof(float));
  int*    row_tok  = (int*)   carve((size_t)2 * T_TOK * sizeof(int));
  int*    tok_slot = (int*)   carve((size_t)2 * T_TOK * sizeof(int));
  int*    counts   = (int*)   carve(64);
  int*    offs     = (int*)   carve(64);

  moe_cvt_x<<<(T_TOK * EMB) / (4 * 256), 256, 0, stream>>>(x, xb);
  moe_router<<<T_TOK / 128, 128, 0, stream>>>(x, router_w, top2w, top2i);
  moe_place<<<1, 256, 0, stream>>>(top2i, row_tok, tok_slot, counts, offs);
  moe_gate_up<<<dim3(T_TOK / BM, FF / BN, NE), 256, 0, stream>>>(
      xb, gate_w, up_w, counts, offs, row_tok, Hbuf);
  moe_down<<<dim3(T_TOK / BM, EMB / BN, NE), 256, 0, stream>>>(
      Hbuf, down_w, counts, offs, Obuf);
  moe_combine<<<T_TOK, 256, 0, stream>>>(tok_slot, top2w, Obuf, out);
}
